// PerChannelFastSSM_28260884807863
// MI455X (gfx1250) — matrix operand order repaired and checked by reading, not compiled, not run on hardware
//
#include <hip/hip_runtime.h>
#include <hip/hip_bf16.h>

// ---------------- problem constants (from reference) ----------------
#define BB      4
#define LF      256      // sequence length (time)
#define LS      128
#define CC      8        // channels
#define DD      8        // lift dim
#define DSTATE  16
#define DCONV   4
#define DI      16       // 2*D
#define NN      (BB*LS)  // 512 sequences per channel
#define TT      LF

typedef _Float16     half16 __attribute__((ext_vector_type(16)));
typedef float        float8 __attribute__((ext_vector_type(8)));
typedef unsigned int v4u    __attribute__((ext_vector_type(4)));
typedef int          v8i    __attribute__((ext_vector_type(8)));
typedef int          v4i    __attribute__((ext_vector_type(4)));

#define DEV __device__ __forceinline__
#define LOG2E 1.4426950408889634f

// fast math: v_rcp_f32 / v_exp_f32 instead of IEEE div chains
DEV float frcp(float x)  { return __builtin_amdgcn_rcpf(x); }
DEV float fexp2(float x) { return __builtin_amdgcn_exp2f(x); }
DEV float sigm(float x)  { return frcp(1.0f + fexp2(-LOG2E * x)); }
DEV float silu(float x)  { return x * sigm(x); }

// WMMA wrapper: D = A(16x32 f16) * B(32x16 f16) + C(16x16 f32)
DEV float8 wmma_f16(half16 a, half16 b, float8 c) {
    return __builtin_amdgcn_wmma_f32_16x16x32_f16(
        /*neg_a=*/false, a, /*neg_b=*/false, b,
        /*c_mod=*/(short)0, c, /*reuse_a=*/false, /*reuse_b=*/false);
}

// A-fragment (16x32, f16) from LDS row-major [row][stride], rows m0..m0+15,
// K columns [0, kmax) valid, rest zero.  CDNA5 16-bit A layout:
//   lanes 0-15: row=lane,  K in {0..7, 16..23}; lanes 16-31: row=lane-16, K in {8..15, 24..31}
DEV half16 load_a_lds(const _Float16* buf, int stride, int m0, int kmax, int lane) {
    half16 a;
    int row   = m0 + (lane & 15);
    int kbase = (lane >> 4) * 8;
#pragma unroll
    for (int i = 0; i < 16; ++i) {
        int vg = i >> 1, e = i & 1;
        int k  = ((vg < 4) ? 0 : 16) + kbase + ((vg & 3) << 1) + e;
        a[i] = (k < kmax) ? buf[row * stride + k] : (_Float16)0.0f;
    }
    return a;
}

// B-fragment (32x16, f16) from a global row-major weight W[n][k] (n rows, ldk stride):
//   B[k][n] = W[n*ldk + k].  CDNA5 16-bit B layout:
//   lane L: column n = L&15 ; K = (i<8 ? 0 : 8) + (L>=16 ? 8 : 0) + i   (i = vector element)
DEV half16 load_b_w(const float* W, int nrows, int kmax, int ldk, int lane) {
    half16 b;
    int n  = lane & 15;
    int kb = (lane >> 4) << 4;
#pragma unroll
    for (int i = 0; i < 16; ++i) {
        int k = (i < 8 ? 0 : 8) + (kb >> 1) + i;
        b[i] = (n < nrows && k < kmax) ? (_Float16)W[n * ldk + k] : (_Float16)0.0f;
    }
    return b;
}

// ---------------------------------------------------------------------------
// Kernel 1: one wave (32 threads) = one (channel c, sequence n) Mamba block.
// x column fetched by the Tensor Data Mover (tensor_load_to_lds, strided 2D
// descriptor); projections via v_wmma_f32_16x16x32_f16 over 16 time-tiles;
// the recurrence in f32 registers with one v_exp_f32 per state update.
// ---------------------------------------------------------------------------
__global__ void __launch_bounds__(32)
mamba_seq_kernel(const float* __restrict__ x,
                 const float* __restrict__ lift_w, const float* __restrict__ lift_b,
                 const float* __restrict__ in_proj_w,
                 const float* __restrict__ conv_w,  const float* __restrict__ conv_b,
                 const float* __restrict__ x_proj_w,
                 const float* __restrict__ dt_w,    const float* __restrict__ dt_b,
                 const float* __restrict__ A_log,   const float* __restrict__ D_skip,
                 const float* __restrict__ out_w,
                 const float* __restrict__ blk_w,   const float* __restrict__ blk_b,
                 float* __restrict__ pooled)        // (C, N, 8)
{
    const int n    = blockIdx.x;        // sequence 0..511  (n = b*LS + ls)
    const int c    = blockIdx.y;        // channel  0..7
    const int lane = threadIdx.x;       // wave32

    // -------- LDS arena (52 KB) with live-range aliasing --------
    __shared__ __align__(16) unsigned char smem[53248];
    _Float16 (*s_z)    [DD]     = (_Float16(*)[DD])    (smem +     0); //  4 KB ph0->1
    _Float16 (*s_xin)  [DI]     = (_Float16(*)[DI])    (smem +  4096); //  8 KB ph1->2
    _Float16 (*s_y)    [DI]     = (_Float16(*)[DI])    (smem +  4096); //        (alias, ph5->6)
    _Float16 (*s_xconv)[DI]     = (_Float16(*)[DI])    (smem + 12288); //  8 KB ph2->5
    _Float16 (*s_gz)   [DI]     = (_Float16(*)[DI])    (smem + 20480); //  8 KB ph1->5
    _Float16 (*s_B)    [DSTATE] = (_Float16(*)[DSTATE])(smem + 28672); //  8 KB ph3->5, stage ph6
    _Float16 (*s_C)    [DSTATE] = (_Float16(*)[DSTATE])(smem + 36864); //  8 KB ph3->5
    _Float16 (*s_dt)   [DI]     = (_Float16(*)[DI])    (smem + 45056); //  8 KB ph4->5
    float*   s_x                = (float*)             (smem + 45056); //  1 KB TDM dest (alias, ph0)

    const float8 cz = {0.f,0.f,0.f,0.f,0.f,0.f,0.f,0.f};
    const int b  = n / LS, ls = n % LS;

    // ================= phase 0a: TDM fetch of x column (strided 2D tile) =================
    // x[b, t, ls, c], t = 0..255  -> 256 rows x 4 bytes, row stride = LS*C elements
#if __has_builtin(__builtin_amdgcn_tensor_load_to_lds)
    {
        unsigned long long ga =
            (unsigned long long)(uintptr_t)&x[(((long)b * LF) * LS + ls) * CC + c];
        unsigned int lds_off = (unsigned int)(uintptr_t)s_x;   // low 32 bits = LDS byte offset
        v4u g0 = { 1u,                                     // count=1 (valid), no gather
                   lds_off,                                // D#.lds_addr
                   (unsigned int)(ga & 0xFFFFFFFFu),       // global_addr[31:0]
                   (unsigned int)((ga >> 32) & 0x1FFFFFFu) // global_addr[56:32]
                     | (2u << 30) };                       // type=2 ("image")
        v8i g1 = { (int)(2u << 16),     // data_size=2 (4 B); wg_mask=0; no pad/iterate
                   (int)(1u << 16),     // tensor_dim0 = 1
                   (int)(256u << 16),   // tensor_dim1 = 256
                   (int)(1u << 16),     // tile_dim0 = 1
                   256,                 // tile_dim1 = 256 (tile_dim2 = 0)
                   1024,                // tensor_dim0_stride = LS*C = 1024 elements
                   0, 0 };
        v4i gz4 = {0, 0, 0, 0};
        v8i gz8 = {0, 0, 0, 0, 0, 0, 0, 0};
        // clang-23 / therock signature: (v4u g0, v8i g1, v4i g2, v4i g3, v8i g4, i32 cpol)
        __builtin_amdgcn_tensor_load_to_lds(g0, g1, gz4, gz4, gz8, 0);
        __builtin_amdgcn_s_wait_tensorcnt(0);
    }
    __syncthreads();
#else
    for (int t = lane; t < TT; t += 32)
        s_x[t] = x[(((long)b * LF + t) * LS + ls) * CC + c];
    __syncthreads();
#endif

    // ================= phase 0b: lift =================
    {
        float lw[DD], lb[DD];
#pragma unroll
        for (int d = 0; d < DD; ++d) { lw[d] = lift_w[c*DD + d]; lb[d] = lift_b[c*DD + d]; }
        for (int t = lane; t < TT; t += 32) {
            float xv = s_x[t];
#pragma unroll
            for (int d = 0; d < DD; ++d)
                s_z[t][d] = (_Float16)(xv * lw[d] + lb[d]);
        }
    }
    __syncthreads();

    // ================= phase 1: in_proj  (256x8)@(8x32) via WMMA =================
    {
        const float* ipw = in_proj_w + (long)c * (2*DI) * DD;   // (32, 8)
        half16 b0 = load_b_w(ipw,          16, DD, DD, lane);   // outputs  0..15 (x half)
        half16 b1 = load_b_w(ipw + 16*DD,  16, DD, DD, lane);   // outputs 16..31 (gate)
        for (int m0 = 0; m0 < TT; m0 += 16) {
            half16 a  = load_a_lds(&s_z[0][0], DD, m0, DD, lane);
            float8 d0 = wmma_f16(a, b0, cz);
            float8 d1 = wmma_f16(a, b1, cz);
            int tb = m0 + ((lane >> 4) << 3);
            int nn = lane & 15;
#pragma unroll
            for (int r = 0; r < 8; ++r) {
                s_xin[tb + r][nn] = (_Float16)d0[r];
                s_gz [tb + r][nn] = (_Float16)silu(d1[r]);
            }
        }
    }
    __syncthreads();

    // ================= phase 2: causal depthwise conv (k=4) + silu =================
    {
        const float* cw = conv_w + (long)c * DI * DCONV;
        const float* cb = conv_b + (long)c * DI;
        for (int idx = lane; idx < TT*DI; idx += 32) {
            int t = idx >> 4, di = idx & 15;
            float acc = cb[di];
#pragma unroll
            for (int k = 0; k < DCONV; ++k) {
                int tt = t + k - (DCONV - 1);
                if (tt >= 0) acc += (float)s_xin[tt][di] * cw[di*DCONV + k];
            }
            s_xconv[t][di] = (_Float16)silu(acc);
        }
    }
    __syncthreads();

    // ================= phase 3: x_proj -> B(t,s), C(t,s) via WMMA =================
    const float* xpw = x_proj_w + (long)c * (1 + 2*DSTATE) * DI;   // (33, 16)
    {
        half16 bBm = load_b_w(xpw + 1*DI,            16, DI, DI, lane);
        half16 bCm = load_b_w(xpw + (1+DSTATE)*DI,   16, DI, DI, lane);
        for (int m0 = 0; m0 < TT; m0 += 16) {
            half16 a  = load_a_lds(&s_xconv[0][0], DI, m0, DI, lane);
            float8 dB = wmma_f16(a, bBm, cz);
            float8 dC = wmma_f16(a, bCm, cz);
            int tb = m0 + ((lane >> 4) << 3);
            int nn = lane & 15;
#pragma unroll
            for (int r = 0; r < 8; ++r) {
                s_B[tb + r][nn] = (_Float16)dB[r];
                s_C[tb + r][nn] = (_Float16)dC[r];
            }
        }
    }
    __syncthreads();

    // ================= phase 4: dt = softplus(dt_lo * dt_w + dt_b)  (rank-1) =================
    {
        for (int t = lane; t < TT; t += 32) {
            float dl = 0.f;
#pragma unroll
            for (int k = 0; k < DI; ++k) dl += (float)s_xconv[t][k] * xpw[k];  // row 0
#pragma unroll
            for (int di = 0; di < DI; ++di) {
                float v  = dl * dt_w[c*DI + di] + dt_b[c*DI + di];
                float sp = (v > 20.f) ? v : log1pf(fexp2(LOG2E * v));
                s_dt[t][di] = (_Float16)sp;
            }
        }
    }
    __syncthreads();

    // ================= phase 5: selective scan (f32 registers, wave-parallel) =================
    {
        const int d  = lane >> 1;          // 0..15  (two lanes per channel-dim)
        const int s0 = (lane & 1) * 8;     // each lane owns 8 of 16 states
        float Av[8];                       // log2(e)*A pre-folded: exp(dt*A) = exp2(dt*Av)
#pragma unroll
        for (int j = 0; j < 8; ++j)
            Av[j] = -LOG2E * __expf(A_log[((long)c*DI + d)*DSTATE + s0 + j]);
        const float dsk = D_skip[c*DI + d];
        float h[8] = {0.f,0.f,0.f,0.f,0.f,0.f,0.f,0.f};

        for (int t = 0; t < TT; ++t) {
            float dtv = (float)s_dt[t][d];
            float xcv = (float)s_xconv[t][d];
            float u   = dtv * xcv;
            float ys  = 0.f;
#pragma unroll
            for (int j = 0; j < 8; ++j) {
                float dA = fexp2(dtv * Av[j]);                         // one v_exp_f32
                h[j] = fmaf(dA, h[j], u * (float)s_B[t][s0 + j]);
                ys   = fmaf(h[j], (float)s_C[t][s0 + j], ys);
            }
            ys += __shfl_xor(ys, 1, 32);                               // reduce state halves
            float y = fmaf(dsk, xcv, ys) * (float)s_gz[t][d];
            s_y[t][d] = (_Float16)y;                                   // both pair lanes write same value
        }
    }
    __syncthreads();

    // ================= phase 6: out_w then blk_w (+bias, silu) via WMMA, mean-pool =================
    float pool = 0.f;
    {
        const float* ow = out_w + (long)c * DD * DI;   // (8, 16)
        const float* bw = blk_w + (long)c * DD * DD;   // (8, 8)
        half16 bO = load_b_w(ow, DD, DI, DI, lane);
        half16 bK = load_b_w(bw, DD, DD, DD, lane);
        float8 cbias = {0.f,0.f,0.f,0.f,0.f,0.f,0.f,0.f};
        {
            int nn = lane & 15;
            float bias = (nn < DD) ? blk_b[c*DD + nn] : 0.f;
#pragma unroll
            for (int r = 0; r < 8; ++r) cbias[r] = bias;
        }
        // stage 1: out = y @ ow^T   -> reuse s_B as the staging buffer (dead after scan)
        for (int m0 = 0; m0 < TT; m0 += 16) {
            half16 a = load_a_lds(&s_y[0][0], DI, m0, DI, lane);
            float8 o = wmma_f16(a, bO, cz);
            int tb = m0 + ((lane >> 4) << 3);
            int nn = lane & 15;
#pragma unroll
            for (int r = 0; r < 8; ++r) s_B[tb + r][nn] = (_Float16)o[r];
        }
        __syncthreads();
        // stage 2: blk = silu(out @ bw^T + bb); accumulate time-mean per column
        for (int m0 = 0; m0 < TT; m0 += 16) {
            half16 a = load_a_lds(&s_B[0][0], DSTATE, m0, DD, lane);
            float8 o = wmma_f16(a, bK, cbias);
#pragma unroll
            for (int r = 0; r < 8; ++r) pool += silu(o[r]);
        }
    }
    pool += __shfl_xor(pool, 16, 32);   // combine M-halves (lanes L and L+16 share column)
    if (lane < DD)
        pooled[((long)c*NN + n)*DD + lane] = pool * (1.0f / TT);
}

// ---------------------------------------------------------------------------
// Kernel 2: LayerNorm over feat (512 rows x 64), feat[n][c*8+d] = pooled[c][n][d]
// ---------------------------------------------------------------------------
__global__ void __launch_bounds__(32)
ln_kernel(const float* __restrict__ pooled,
          const float* __restrict__ ln_g, const float* __restrict__ ln_b,
          float* __restrict__ out)
{
    const int n    = blockIdx.x;
    const int lane = threadIdx.x;
    float vals[2];
#pragma unroll
    for (int p = 0; p < 2; ++p) {
        int j  = lane + p*32;            // feature index 0..63
        int cc = j >> 3, dd = j & 7;
        vals[p] = pooled[((long)cc*NN + n)*DD + dd];
    }
    float s = vals[0] + vals[1];
#pragma unroll
    for (int o = 16; o > 0; o >>= 1) s += __shfl_xor(s, o, 32);
    float mu = s * (1.0f/64.0f);
    float q = (vals[0]-mu)*(vals[0]-mu) + (vals[1]-mu)*(vals[1]-mu);
#pragma unroll
    for (int o = 16; o > 0; o >>= 1) q += __shfl_xor(q, o, 32);
    float rstd = rsqrtf(q * (1.0f/64.0f) + 1e-5f);
#pragma unroll
    for (int p = 0; p < 2; ++p) {
        int j = lane + p*32;
        out[(long)n*64 + j] = (vals[p]-mu)*rstd*ln_g[j] + ln_b[j];
    }
}

extern "C" void kernel_launch(void* const* d_in, const int* in_sizes, int n_in,
                              void* d_out, int out_size, void* d_ws, size_t ws_size,
                              hipStream_t stream) {
    const float* x         = (const float*)d_in[0];
    const float* lift_w    = (const float*)d_in[1];
    const float* lift_b    = (const float*)d_in[2];
    const float* in_proj_w = (const float*)d_in[3];
    const float* conv_w    = (const float*)d_in[4];
    const float* conv_b    = (const float*)d_in[5];
    const float* x_proj_w  = (const float*)d_in[6];
    const float* dt_w      = (const float*)d_in[7];
    const float* dt_b      = (const float*)d_in[8];
    const float* A_log     = (const float*)d_in[9];
    const float* D_skip    = (const float*)d_in[10];
    const float* out_w     = (const float*)d_in[11];
    const float* blk_w     = (const float*)d_in[12];
    const float* blk_b     = (const float*)d_in[13];
    const float* ln_g      = (const float*)d_in[14];
    const float* ln_b      = (const float*)d_in[15];

    float* pooled = (float*)d_ws;                 // C*N*8 f32 = 128 KB
    float* out    = (float*)d_out;                // (4, 128, 64) f32

    dim3 grid1(NN, CC);
    mamba_seq_kernel<<<grid1, 32, 0, stream>>>(
        x, lift_w, lift_b, in_proj_w, conv_w, conv_b, x_proj_w,
        dt_w, dt_b, A_log, D_skip, out_w, blk_w, blk_b, pooled);

    ln_kernel<<<NN, 32, 0, stream>>>(pooled, ln_g, ln_b, out);
}
